// Decoder_10411000725932
// MI455X (gfx1250) — matrix operand order repaired and checked by reading, not compiled, not run on hardware
//
#include <hip/hip_runtime.h>
#include <hip/hip_bf16.h>

// ---------------------------------------------------------------------------
// Social-GAN decoder on gfx1250: SEQ=12 sequential steps, B=1536 (64 grp x 24),
// E=64, H=128, BOTTLE=1024.  All GEMMs use f16-in / f32-acc WMMA
// (v_wmma_f32_16x16x32_f16); LDS staging uses GLOBAL_LOAD_ASYNC_TO_LDS_B128
// (ASYNCcnt) when the toolchain exposes the builtin.
// ---------------------------------------------------------------------------

#define SEQ      12
#define NB       1536
#define NG       64
#define NP       24
#define EDIM     64
#define HDIM     128
#define BOT      1024

typedef __attribute__((ext_vector_type(16))) _Float16 v16h;
typedef __attribute__((ext_vector_type(8)))  float    v8f;
typedef __attribute__((ext_vector_type(4)))  int      v4i;

#define GLOBAL_AS __attribute__((address_space(1)))
#define LDS_AS    __attribute__((address_space(3)))

#if defined(__has_builtin)
#if __has_builtin(__builtin_amdgcn_global_load_async_to_lds_b128)
#define HAVE_ASYNC_LDS 1
#endif
#endif

// 16-byte global -> LDS copy; async (ASYNCcnt) when available.
__device__ __forceinline__ void async_copy16(void* lds_dst, const void* gsrc) {
#if defined(HAVE_ASYNC_LDS)
  __builtin_amdgcn_global_load_async_to_lds_b128(
      (GLOBAL_AS v4i*)gsrc, (LDS_AS v4i*)lds_dst, 0, 0);
#else
  *(uint4*)lds_dst = *(const uint4*)gsrc;
#endif
}

__device__ __forceinline__ void wait_async_copies() {
#if defined(HAVE_ASYNC_LDS)
#if __has_builtin(__builtin_amdgcn_s_wait_asynccnt)
  __builtin_amdgcn_s_wait_asynccnt(0);
#else
  asm volatile("s_wait_asynccnt 0x0" ::: "memory");
#endif
#endif
}

__device__ __forceinline__ v8f wmma_f16(v16h a, v16h b, v8f c) {
  // 8 args: (neg_a, A, neg_b, B, c_mod, C, reuse_a, reuse_b)
  return __builtin_amdgcn_wmma_f32_16x16x32_f16(false, a, false, b, (short)0, c,
                                                false, false);
}

// B-fragment (32x16, f16) from row-major W[N][K]:  N = lane%16,
// K = e + 16*(lane/16)  -> 16 contiguous halves per lane (2x b128).
__device__ __forceinline__ v16h ld_bfrag(const _Float16* __restrict__ W, int K,
                                         int n0, int k0, int lane) {
  const _Float16* p = W + (size_t)(n0 + (lane & 15)) * K + k0 + ((lane >> 4) << 4);
  return *(const v16h*)p;
}

// A-fragment (16x32, f16) from an LDS tile with leading dim ldk.
// Row M = lane%16; element pair v holds K = 2v + 16*(lane/16).
__device__ __forceinline__ v16h ld_afrag_lds(const _Float16* tile, int ldk,
                                             int k0, int lane) {
  const _Float16* p = tile + (lane & 15) * ldk + k0;
  const int hi = (lane >> 4) << 4;   // 0 or 16
  v16h a;
#pragma unroll
  for (int v = 0; v < 8; ++v) {
    int kk = 2 * v + hi;
    a[2 * v]     = p[kk];
    a[2 * v + 1] = p[kk + 1];
  }
  return a;
}

// ---------------------------------------------------------------------------
// Generic fused GEMM:  Y = act(X[M,K] * W[N,K]^T + bias)
// 64x64 tile per 256-thread block (8 waves); K staged 64-wide into LDS via
// async copies (2x b128 per thread per chunk); wave w -> M-tile (w&3),
// N-tiles at (w>>2)*32 + {0,16}.  Optional f32 and/or f16 outputs.
// K must be a multiple of 64.
// ---------------------------------------------------------------------------
__global__ void __launch_bounds__(256) k_gemm(
    const _Float16* __restrict__ X, const _Float16* __restrict__ W,
    const float* __restrict__ bias, float* __restrict__ Yf,
    _Float16* __restrict__ Yh, int M, int N, int K, int relu) {
  __shared__ __align__(16) _Float16 sA[64 * 64];   // 8 KB
  const int tid  = threadIdx.x;
  const int lane = tid & 31;
  const int w    = tid >> 5;
  const int mt   = w & 3;
  const int ntp  = w >> 2;
  const int mblk = blockIdx.y * 64;
  const int nblk = blockIdx.x * 64;
  const int n0   = nblk + ntp * 32;

  v8f acc0 = {}, acc1 = {};
  for (int k0 = 0; k0 < K; k0 += 64) {
    // stage X[mblk..mblk+63][k0..k0+63] -> LDS (2x 16B per thread)
#pragma unroll
    for (int u2 = 0; u2 < 2; ++u2) {
      const int u = tid * 2 + u2;      // 0..511 : 64 rows x 8 16B-units
      const int r = u >> 3;
      const int c = (u & 7) * 8;       // half-column
      async_copy16(&sA[r * 64 + c], X + (size_t)(mblk + r) * K + k0 + c);
    }
    if (k0 + 64 < K)  // prefetch next chunk of this wave's B rows into L2/L0
      __builtin_prefetch(W + (size_t)(n0 + (lane & 15)) * K + k0 + 64, 0, 3);
    wait_async_copies();
    __syncthreads();
#pragma unroll
    for (int kk = 0; kk < 64; kk += 32) {
      v16h A  = ld_afrag_lds(&sA[mt * 16 * 64], 64, kk, lane);
      v16h B0 = ld_bfrag(W, K, n0,      k0 + kk, lane);
      v16h B1 = ld_bfrag(W, K, n0 + 16, k0 + kk, lane);
      acc0 = wmma_f16(A, B0, acc0);
      acc1 = wmma_f16(A, B1, acc1);
    }
    __syncthreads();
  }
  const int m0 = mblk + mt * 16 + ((lane >> 4) << 3);
  const int nA = n0 + (lane & 15);
  const int nB = nA + 16;
  const float bA = bias[nA], bB = bias[nB];
#pragma unroll
  for (int r = 0; r < 8; ++r) {
    const int m = m0 + r;
    float v0 = acc0[r] + bA;
    float v1 = acc1[r] + bB;
    if (relu) { v0 = fmaxf(v0, 0.f); v1 = fmaxf(v1, 0.f); }
    if (Yf) { Yf[(size_t)m * N + nA] = v0; Yf[(size_t)m * N + nB] = v1; }
    if (Yh) { Yh[(size_t)m * N + nA] = (_Float16)v0;
              Yh[(size_t)m * N + nB] = (_Float16)v1; }
  }
}

// ---------------------------------------------------------------------------
// Fused pool_net: one block per (group g, pedestrian i).
//   X[j,0:64]   = Linear2->64( normalize(pos_j - pos_i) )   (computed)
//   X[j,64:192] = h_j                                       (async from h16)
//   Y1 = relu(X @ Wp1^T + bp1)   [24x512 padded to 32 rows, LDS]
//   Y2 = Y1 @ Wp2^T              [24x1024, registers only]
//   out[g*24+i, n] = relu(max_j Y2[j,n] + bp2[n])
// ---------------------------------------------------------------------------
__global__ void __launch_bounds__(256) k_pool(
    const float* __restrict__ cur, const _Float16* __restrict__ h16,
    const _Float16* __restrict__ Wp1, const float* __restrict__ bp1,
    const _Float16* __restrict__ Wp2, const float* __restrict__ bp2,
    const float* __restrict__ Wpe, const float* __restrict__ bpe,
    float* __restrict__ pool_out) {
  __shared__ __align__(16) _Float16 sX[32 * 192];    // 12 KB
  __shared__ __align__(16) _Float16 sY1[32 * 512];   // 32 KB
  const int tid = threadIdx.x, lane = tid & 31, w = tid >> 5;
  const int gi = blockIdx.x;              // 0..1535
  const int g = gi / NP, i = gi % NP;
  const float pix = cur[(g * NP + i) * 2 + 0];
  const float piy = cur[(g * NP + i) * 2 + 1];

  // phase 0a: async copy hidden rows (24 rows x 256B = 384 b128 units)
  for (int u = tid; u < NP * 16; u += 256) {
    const int j = u >> 4;
    const int s = (u & 15) * 8;          // half offset within the 128-wide row
    async_copy16(&sX[j * 192 + EDIM + s],
                 h16 + (size_t)(g * NP + j) * HDIM + s);
  }
  // phase 0b: embedding columns for valid rows
  for (int idx = tid; idx < NP * EDIM; idx += 256) {
    const int j = idx / EDIM, col = idx % EDIM;
    float rx = cur[(g * NP + j) * 2 + 0] - pix;
    float ry = cur[(g * NP + j) * 2 + 1] - piy;
    float d  = fmaxf(sqrtf(rx * rx + ry * ry), 1e-12f);
    rx /= d; ry /= d;
    sX[j * 192 + col] =
        (_Float16)(rx * Wpe[col * 2 + 0] + ry * Wpe[col * 2 + 1] + bpe[col]);
  }
  // phase 0c: zero padding rows 24..31
  for (int idx = tid; idx < 8 * 192; idx += 256)
    sX[NP * 192 + idx] = (_Float16)0.f;
  wait_async_copies();
  __syncthreads();

  // phase 1: Y1[32,512], wave w owns N-tiles 4w..4w+3
  {
    v8f acc[2][4] = {};
    for (int kt = 0; kt < 6; ++kt) {
      const int k0 = kt * 32;
      v16h A0 = ld_afrag_lds(&sX[0],        192, k0, lane);
      v16h A1 = ld_afrag_lds(&sX[16 * 192], 192, k0, lane);
#pragma unroll
      for (int q = 0; q < 4; ++q) {
        v16h B = ld_bfrag(Wp1, 192, (w * 4 + q) * 16, k0, lane);
        acc[0][q] = wmma_f16(A0, B, acc[0][q]);
        acc[1][q] = wmma_f16(A1, B, acc[1][q]);
      }
    }
    const int hi = lane >> 4, nn = lane & 15;
#pragma unroll
    for (int q = 0; q < 4; ++q) {
      const int n = (w * 4 + q) * 16 + nn;
      const float b = bp1[n];
#pragma unroll
      for (int mt = 0; mt < 2; ++mt)
#pragma unroll
        for (int r = 0; r < 8; ++r) {
          const int m = mt * 16 + r + (hi << 3);
          sY1[m * 512 + n] = (_Float16)fmaxf(acc[mt][q][r] + b, 0.f);
        }
    }
  }
  __syncthreads();

  // phase 2: Y2[32,1024] in registers + masked max over valid rows (j<24)
  {
    v8f acc[2][8] = {};
    for (int kt = 0; kt < 16; ++kt) {
      const int k0 = kt * 32;
      v16h A0 = ld_afrag_lds(&sY1[0],        512, k0, lane);
      v16h A1 = ld_afrag_lds(&sY1[16 * 512], 512, k0, lane);
#pragma unroll
      for (int q = 0; q < 8; ++q) {
        v16h B = ld_bfrag(Wp2, 512, (w * 8 + q) * 16, k0, lane);
        acc[0][q] = wmma_f16(A0, B, acc[0][q]);
        acc[1][q] = wmma_f16(A1, B, acc[1][q]);
      }
    }
    const int hi = lane >> 4, nn = lane & 15;
#pragma unroll
    for (int q = 0; q < 8; ++q) {
      float vmax = -3.0e38f;
#pragma unroll
      for (int r = 0; r < 8; ++r) vmax = fmaxf(vmax, acc[0][q][r]);  // rows 0..15
      if (!hi) {                       // mt=1, lanes<16 -> rows 16..23 (valid)
#pragma unroll
        for (int r = 0; r < 8; ++r) vmax = fmaxf(vmax, acc[1][q][r]);
      }                                // mt=1, lanes>=16 -> rows 24..31 (pad)
      const float other = __shfl_xor(vmax, 16, 32);
      vmax = fmaxf(vmax, other);
      if (!hi) {
        const int n = (w * 8 + q) * 16 + nn;
        pool_out[(size_t)gi * BOT + n] = fmaxf(vmax + bp2[n], 0.f);
      }
    }
  }
}

// ------------------------- elementwise helpers -----------------------------

__global__ void k_convert(const float* __restrict__ s, _Float16* __restrict__ d, int n) {
  int i = blockIdx.x * blockDim.x + threadIdx.x;
  if (i < n) d[i] = (_Float16)s[i];
}

__global__ void k_build_lstm_w(const float* __restrict__ Wih, const float* __restrict__ Whh,
                               const float* __restrict__ bih, const float* __restrict__ bhh,
                               _Float16* __restrict__ Wc, float* __restrict__ bc) {
  int idx = blockIdx.x * blockDim.x + threadIdx.x;
  if (idx >= 512 * 192) return;
  const int n = idx / 192, k = idx % 192;
  const float v = (k < 64) ? Wih[n * 64 + k] : Whh[n * 128 + (k - 64)];
  Wc[idx] = (_Float16)v;
  if (k == 0) bc[n] = bih[n] + bhh[n];
}

__global__ void k_init(const float* __restrict__ lpin, const float* __restrict__ relin,
                       const float* __restrict__ h0, const float* __restrict__ c0,
                       float* __restrict__ lp, float* __restrict__ rel,
                       float* __restrict__ h, float* __restrict__ c,
                       _Float16* __restrict__ h16) {
  int idx = blockIdx.x * blockDim.x + threadIdx.x;
  if (idx < NB * HDIM) {
    h[idx] = h0[idx]; c[idx] = c0[idx]; h16[idx] = (_Float16)h0[idx];
  }
  if (idx < NB * 2) { lp[idx] = lpin[idx]; rel[idx] = relin[idx]; }
}

// X_lstm[b] = [ emb(rel_b) | h16_b ]  (f16, 192 cols)
__global__ void k_embed_pack(const float* __restrict__ rel, const _Float16* __restrict__ h16,
                             const float* __restrict__ We, const float* __restrict__ be,
                             _Float16* __restrict__ Xl) {
  const int b = blockIdx.x, c = threadIdx.x;   // 192 threads
  _Float16 v;
  if (c < EDIM) {
    v = (_Float16)(rel[b * 2] * We[c * 2] + rel[b * 2 + 1] * We[c * 2 + 1] + be[c]);
  } else {
    v = h16[(size_t)b * HDIM + (c - EDIM)];
  }
  Xl[(size_t)b * 192 + c] = v;
}

__global__ void k_lstm_update(const float* __restrict__ gates,
                              float* __restrict__ c, float* __restrict__ h,
                              _Float16* __restrict__ h16) {
  int idx = blockIdx.x * blockDim.x + threadIdx.x;
  if (idx >= NB * HDIM) return;
  const int b = idx >> 7, d = idx & 127;
  const float* gp = gates + (size_t)b * 512;
  const float si = 1.f / (1.f + __expf(-gp[d]));
  const float sf = 1.f / (1.f + __expf(-gp[128 + d]));
  const float cg = tanhf(gp[256 + d]);
  const float so = 1.f / (1.f + __expf(-gp[384 + d]));
  const float cn = sf * c[idx] + si * cg;
  const float hv = so * tanhf(cn);
  c[idx] = cn;
  h[idx] = hv;
  h16[idx] = (_Float16)hv;
}

// rel = h @ W_pos^T + b_pos ; lp += rel ; store rel to d_out[t]
__global__ void k_pos(const float* __restrict__ h, const float* __restrict__ Wpos,
                      const float* __restrict__ bpos, float* __restrict__ lp,
                      float* __restrict__ rel, float* __restrict__ out_rels) {
  int b = blockIdx.x * blockDim.x + threadIdx.x;
  if (b >= NB) return;
  const float* hp = h + (size_t)b * HDIM;
  float r0 = bpos[0], r1 = bpos[1];
  for (int k = 0; k < HDIM; ++k) {
    const float hv = hp[k];
    r0 += hv * Wpos[k];
    r1 += hv * Wpos[HDIM + k];
  }
  rel[b * 2 + 0] = r0;          rel[b * 2 + 1] = r1;
  lp[b * 2 + 0] += r0;          lp[b * 2 + 1] += r1;
  out_rels[b * 2 + 0] = r0;     out_rels[b * 2 + 1] = r1;
}

__global__ void k_pack_m1(const float* __restrict__ h, const float* __restrict__ pool,
                          _Float16* __restrict__ Xm1) {
  int idx = blockIdx.x * blockDim.x + threadIdx.x;
  if (idx >= NB * (HDIM + BOT)) return;
  const int b = idx / (HDIM + BOT), c = idx % (HDIM + BOT);
  const float v = (c < HDIM) ? h[(size_t)b * HDIM + c]
                             : pool[(size_t)b * BOT + (c - HDIM)];
  Xm1[idx] = (_Float16)v;
}

__global__ void k_final(const float* __restrict__ h, float* __restrict__ out) {
  int idx = blockIdx.x * blockDim.x + threadIdx.x;
  if (idx < NB * HDIM) out[idx] = h[idx];
}

// ---------------------------------------------------------------------------
extern "C" void kernel_launch(void* const* d_in, const int* in_sizes, int n_in,
                              void* d_out, int out_size, void* d_ws, size_t ws_size,
                              hipStream_t stream) {
  const float* last_pos     = (const float*)d_in[0];
  const float* last_pos_rel = (const float*)d_in[1];
  const float* h0           = (const float*)d_in[2];
  const float* c0           = (const float*)d_in[3];
  /* d_in[4] = seq_start_end (contiguous equal groups -> unused) */
  const float* W_emb  = (const float*)d_in[5];
  const float* b_emb  = (const float*)d_in[6];
  const float* W_ih   = (const float*)d_in[7];
  const float* W_hh   = (const float*)d_in[8];
  const float* b_ih   = (const float*)d_in[9];
  const float* b_hh   = (const float*)d_in[10];
  const float* W_pos  = (const float*)d_in[11];
  const float* b_pos  = (const float*)d_in[12];
  const float* Wp_emb = (const float*)d_in[13];
  const float* bp_emb = (const float*)d_in[14];
  const float* Wp1    = (const float*)d_in[15];
  const float* bp1    = (const float*)d_in[16];
  const float* Wp2    = (const float*)d_in[17];
  const float* bp2    = (const float*)d_in[18];
  const float* Wm1    = (const float*)d_in[19];
  const float* bm1    = (const float*)d_in[20];
  const float* Wm2    = (const float*)d_in[21];
  const float* bm2    = (const float*)d_in[22];
  float* out = (float*)d_out;

  // workspace carve-out (~24 MB total)
  char* ws = (char*)d_ws;
  size_t off = 0;
  auto carve = [&](size_t bytes) -> char* {
    char* p = ws + off;
    off = (off + bytes + 255) & ~(size_t)255;
    return p;
  };
  _Float16* Wc_h  = (_Float16*)carve(512 * 192 * 2);
  float*    bc    = (float*)   carve(512 * 4);
  _Float16* Wp1h  = (_Float16*)carve(512 * 192 * 2);
  _Float16* Wp2h  = (_Float16*)carve((size_t)BOT * 512 * 2);
  _Float16* Wm1h  = (_Float16*)carve((size_t)BOT * (HDIM + BOT) * 2);
  _Float16* Wm2h  = (_Float16*)carve((size_t)HDIM * BOT * 2);
  float*    hbuf  = (float*)   carve((size_t)NB * HDIM * 4);
  float*    cbuf  = (float*)   carve((size_t)NB * HDIM * 4);
  _Float16* h16   = (_Float16*)carve((size_t)NB * HDIM * 2);
  float*    lp    = (float*)   carve((size_t)NB * 2 * 4);
  float*    rel   = (float*)   carve((size_t)NB * 2 * 4);
  _Float16* Xl    = (_Float16*)carve((size_t)NB * 192 * 2);
  float*    gates = (float*)   carve((size_t)NB * 512 * 4);
  float*    pool  = (float*)   carve((size_t)NB * BOT * 4);
  _Float16* Xm1   = (_Float16*)carve((size_t)NB * (HDIM + BOT) * 2);
  _Float16* Xm2   = (_Float16*)carve((size_t)NB * BOT * 2);
  (void)ws_size; (void)in_sizes; (void)n_in; (void)out_size;

  // one-time (per launch) weight conversion to f16
  k_convert<<<(512 * 192 + 255) / 256, 256, 0, stream>>>(Wp1, Wp1h, 512 * 192);
  k_convert<<<(BOT * 512 + 255) / 256, 256, 0, stream>>>(Wp2, Wp2h, BOT * 512);
  k_convert<<<(BOT * (HDIM + BOT) + 255) / 256, 256, 0, stream>>>(Wm1, Wm1h, BOT * (HDIM + BOT));
  k_convert<<<(HDIM * BOT + 255) / 256, 256, 0, stream>>>(Wm2, Wm2h, HDIM * BOT);
  k_build_lstm_w<<<(512 * 192 + 255) / 256, 256, 0, stream>>>(W_ih, W_hh, b_ih, b_hh, Wc_h, bc);
  k_init<<<(NB * HDIM + 255) / 256, 256, 0, stream>>>(last_pos, last_pos_rel, h0, c0,
                                                      lp, rel, hbuf, cbuf, h16);

  for (int t = 0; t < SEQ; ++t) {
    // X_lstm = [emb(rel) | h] f16
    k_embed_pack<<<NB, 192, 0, stream>>>(rel, h16, W_emb, b_emb, Xl);
    // gates = X_lstm @ Wc^T + (b_ih + b_hh)      (K=192 = 3x64)
    k_gemm<<<dim3(512 / 64, NB / 64), 256, 0, stream>>>(
        Xl, Wc_h, bc, gates, nullptr, NB, 512, 192, 0);
    k_lstm_update<<<(NB * HDIM + 255) / 256, 256, 0, stream>>>(gates, cbuf, hbuf, h16);
    // rel_pos, curr_pos(=lp), write rels[t]
    k_pos<<<(NB + 255) / 256, 256, 0, stream>>>(hbuf, W_pos, b_pos, lp, rel,
                                                out + (size_t)t * NB * 2);
    // pool_net (heavy fused WMMA kernel)
    k_pool<<<NB, 256, 0, stream>>>(lp, h16, Wp1h, bp1, Wp2h, bp2,
                                   Wp_emb, bp_emb, pool);
    // dh = [h | pool] -> relu GEMMs; second GEMM refreshes h (f32 + f16)
    k_pack_m1<<<(NB * (HDIM + BOT) + 255) / 256, 256, 0, stream>>>(hbuf, pool, Xm1);
    k_gemm<<<dim3(BOT / 64, NB / 64), 256, 0, stream>>>(
        Xm1, Wm1h, bm1, nullptr, Xm2, NB, BOT, HDIM + BOT, 1);
    k_gemm<<<dim3(HDIM / 64, NB / 64), 256, 0, stream>>>(
        Xm2, Wm2h, bm2, hbuf, h16, NB, HDIM, BOT, 1);
  }
  k_final<<<(NB * HDIM + 255) / 256, 256, 0, stream>>>(hbuf, out + (size_t)SEQ * NB * 2);
}
